// fusion_loss_mef_45440753991818
// MI455X (gfx1250) — matrix operand order repaired and checked by reading, not compiled, not run on hardware
//
#include <hip/hip_runtime.h>
#include <math.h>

#define HH 512
#define WW 512
#define HWSZ (HH * WW)        // 262144 = 1<<18
#define NBATCH 4
#define NTOT (NBATCH * HWSZ)  // 1048576

typedef __attribute__((ext_vector_type(16))) _Float16 v16h;
typedef __attribute__((ext_vector_type(8)))  float    v8f;

union U16h { v16h v; _Float16 s[16]; };
union U8f  { v8f  v; float    s[8];  };

// ---------------- helpers ----------------

__device__ __forceinline__ unsigned ordKey(float f) {
    unsigned u = __float_as_uint(f);
    return (u & 0x80000000u) ? ~u : (u | 0x80000000u);
}
__device__ __forceinline__ float ordToF(unsigned u) {
    unsigned v = (u & 0x80000000u) ? (u ^ 0x80000000u) : ~u;
    return __uint_as_float(v);
}

__device__ __forceinline__ float blockReduce256(float v, float* s) {
    int tid = threadIdx.x;
    s[tid] = v;
    __syncthreads();
    for (int off = 128; off > 0; off >>= 1) {
        if (tid < off) s[tid] += s[tid + off];
        __syncthreads();
    }
    float r = s[0];
    __syncthreads();
    return r;
}

__device__ __forceinline__ int clampi(int v, int lo, int hi) {
    return min(max(v, lo), hi);
}

// branch-free zero-padded load (address always legal, value selected)
__device__ __forceinline__ float ldc(const float* __restrict__ p, int y, int x) {
    int yc = clampi(y, 0, HH - 1), xc = clampi(x, 0, WW - 1);
    float v = p[yc * WW + xc];
    bool inb = ((unsigned)y < (unsigned)HH) & ((unsigned)x < (unsigned)WW);
    return inb ? v : 0.f;
}

__device__ __forceinline__ float sobelAt(const float* __restrict__ p, int y, int x) {
    float a00 = ldc(p, y - 1, x - 1), a01 = ldc(p, y - 1, x), a02 = ldc(p, y - 1, x + 1);
    float a10 = ldc(p, y,     x - 1),                         a12 = ldc(p, y,     x + 1);
    float a20 = ldc(p, y + 1, x - 1), a21 = ldc(p, y + 1, x), a22 = ldc(p, y + 1, x + 1);
    float gx = (a02 + 2.f * a12 + a22) - (a00 + 2.f * a10 + a20);
    float gy = (a00 + 2.f * a01 + a02) - (a20 + 2.f * a21 + a22);
    return fabsf(gx) + fabsf(gy);
}

// ---------------- zero scratch scalars ----------------

__global__ void zeroK(float* __restrict__ p, int n) {
    int i = blockIdx.x * blockDim.x + threadIdx.x;
    if (i < n) p[i] = 0.f;
}

// ---------------- exact quantile via radix select ----------------

__device__ float radixSelect(const float* __restrict__ src, unsigned k,
                             unsigned* hist, unsigned* sPre, unsigned* sRem, int tid) {
    unsigned prefix = 0;
    unsigned remain = k;
    for (int shift = 24; shift >= 0; shift -= 8) {
        hist[tid] = 0;          // blockDim == 256
        __syncthreads();
        unsigned hmask = (shift == 24) ? 0u : (0xFFFFFFFFu << (shift + 8));
        for (int i = tid; i < HWSZ; i += 256) {
            unsigned u = ordKey(src[i]);
            if ((u & hmask) == (prefix & hmask))
                atomicAdd(&hist[(u >> shift) & 255u], 1u);
        }
        __syncthreads();
        if (tid == 0) {
            unsigned cum = 0;
            unsigned pre = prefix, rem = remain;
            for (int b = 0; b < 256; ++b) {
                unsigned c = hist[b];
                if (cum + c > remain) { pre = prefix | ((unsigned)b << shift); rem = remain - cum; break; }
                cum += c;
            }
            *sPre = pre; *sRem = rem;
        }
        __syncthreads();
        prefix = *sPre; remain = *sRem;
        __syncthreads();
    }
    return ordToF(prefix);
}

__global__ void quantK(const float* __restrict__ A, const float* __restrict__ Bp,
                       const float* __restrict__ S, int phase, float* __restrict__ quant) {
    __shared__ unsigned hist[256];
    __shared__ unsigned sPre, sRem;
    int tid = threadIdx.x;
    int img = blockIdx.y;
    int task = blockIdx.x;
    const float* src; double q; int slot;
    if (phase == 0) {
        switch (task) {
            case 0:  src = A;  q = 0.70; slot = 0; break;
            case 1:  src = A;  q = 0.97; slot = 1; break;
            case 2:  src = Bp; q = 0.40; slot = 2; break;
            case 3:  src = Bp; q = 0.85; slot = 3; break;
            default: src = Bp; q = 0.97; slot = 4; break;
        }
    } else {
        if (task == 0) { src = S; q = 0.90; slot = 5; }
        else           { src = S; q = 0.97; slot = 6; }
    }
    src += (size_t)img * HWSZ;
    double kD = q * (double)(HWSZ - 1);
    unsigned k0 = (unsigned)kD;
    float frac = (float)(kD - (double)k0);
    float v0 = radixSelect(src, k0,     hist, &sPre, &sRem, tid);
    float v1 = radixSelect(src, k0 + 1, hist, &sPre, &sRem, tid);
    if (tid == 0) quant[img * 8 + slot] = v0 + frac * (v1 - v0);
}

// ---------------- separable pooling (min / max / sum), branch-free borders ----------------

template<int OP, int HORIZ>
__global__ void poolK(const float* __restrict__ src, float* __restrict__ dst,
                      int radius, float scale, int clip01) {
    int idx = blockIdx.x * blockDim.x + threadIdx.x;
    if (idx >= NTOT) return;
    int rem = idx & (HWSZ - 1);
    int y = rem >> 9, x = rem & (WW - 1);
    const float* base = src + (idx - rem);
    const float ident = (OP == 0) ? INFINITY : (OP == 1) ? -INFINITY : 0.f;
    float acc = ident;
    for (int d = -radius; d <= radius; ++d) {
        int yy = HORIZ ? y : (y + d);
        int xx = HORIZ ? (x + d) : x;
        bool inb = ((unsigned)yy < (unsigned)HH) & ((unsigned)xx < (unsigned)WW);
        int yyc = HORIZ ? y : clampi(yy, 0, HH - 1);
        int xxc = HORIZ ? clampi(xx, 0, WW - 1) : x;
        float v = base[yyc * WW + xxc];
        v = inb ? v : ident;
        acc = (OP == 0) ? fminf(acc, v) : (OP == 1) ? fmaxf(acc, v) : (acc + v);
    }
    acc *= scale;
    if (clip01) acc = fminf(fmaxf(acc, 0.f), 1.f);
    dst[idx] = acc;
}

__global__ void salUpdateK(const float* __restrict__ A, const float* __restrict__ open,
                           float* __restrict__ sal, int first) {
    int idx = blockIdx.x * blockDim.x + threadIdx.x;
    if (idx >= NTOT) return;
    float v = A[idx] - open[idx];
    sal[idx] = first ? v : fmaxf(sal[idx], v);
}

__global__ void chunkSumK(const float* __restrict__ src, float* __restrict__ Pout) {
    __shared__ float sred[256];
    int blk = blockIdx.x, tid = threadIdx.x;
    int base = blk * 2048;
    float a = 0.f;
    for (int i = tid; i < 2048; i += 256) a += src[base + i];
    float t = blockReduce256(a, sred);
    if (tid == 0) Pout[blk] = t;
}

__global__ void salMeanK(const float* __restrict__ P, float* __restrict__ salmean) {
    if (threadIdx.x != 0 || blockIdx.x != 0) return;
    for (int i = 0; i < NBATCH; ++i) {
        float s = 0.f;
        for (int j = 0; j < 128; ++j) s += P[i * 128 + j];
        salmean[i] = s / (float)HWSZ;
    }
}

__global__ void salNormK(float* __restrict__ sal, const float* __restrict__ salmean) {
    int idx = blockIdx.x * blockDim.x + threadIdx.x;
    if (idx >= NTOT) return;
    int img = idx >> 18;
    sal[idx] = sal[idx] / (salmean[img] + 1e-6f);
}

// ---------------- M / M_light ----------------

__global__ void computeMK(const float* __restrict__ sal, const float* __restrict__ A,
                          const float* __restrict__ Bp, const float* __restrict__ quant,
                          float* __restrict__ M, float* __restrict__ Ml) {
    int idx = blockIdx.x * blockDim.x + threadIdx.x;
    if (idx >= NTOT) return;
    int img = idx >> 18;
    float qs90 = quant[img * 8 + 5], qs97 = quant[img * 8 + 6];
    float qa97 = quant[img * 8 + 1], qb97 = quant[img * 8 + 4];
    float s = sal[idx];
    float z = 20.f * (s - qs90);
    M[idx] = 1.f / (1.f + expf(-z));
    Ml[idx] = (s > qs97 && A[idx] > qa97 && Bp[idx] > qb97) ? 1.f : 0.f;
}

// ---------------- L1 intensity loss ----------------

__global__ void l1K(const float* __restrict__ A, const float* __restrict__ Bp,
                    const float* __restrict__ F, const float* __restrict__ M,
                    const float* __restrict__ quant, float* __restrict__ P) {
    __shared__ float sred[256];
    int blk = blockIdx.x, tid = threadIdx.x;
    int base = blk * 2048;
    int img = blk >> 7;
    float thr_ir  = fminf(fmaxf(quant[img * 8 + 0], 0.45f), 0.85f);
    float thr_vis = fminf(fmaxf(quant[img * 8 + 2], 0.30f), 0.70f);
    float thr_sup = fminf(fmaxf(quant[img * 8 + 3], 0.75f), 0.98f);
    float acc = 0.f;
    for (int i = tid; i < 2048; i += 256) {
        int idx = base + i;
        float a = A[idx], b = Bp[idx], f = F[idx];
        float imax = fmaxf(a, b), imin = fminf(a, b);
        float salient = (b > thr_vis) ? imin : imax;
        salient = ((b > thr_sup) && (a > thr_ir)) ? imax : salient;
        float m = M[idx];
        float target = m * salient + (1.f - m) * imax;
        acc += fabsf(f - target);
    }
    float t = blockReduce256(acc, sred);
    if (tid == 0) P[1 * 512 + blk] = t;
}

// ---------------- Sobel gradient loss ----------------

__global__ void sobelK(const float* __restrict__ A, const float* __restrict__ Bp,
                       const float* __restrict__ F, float* __restrict__ P) {
    __shared__ float sred[256];
    int blk = blockIdx.x, tid = threadIdx.x;
    int base = blk * 2048;
    float acc = 0.f;
    for (int i = tid; i < 2048; i += 256) {
        int idx = base + i;
        int rem = idx & (HWSZ - 1);
        int y = rem >> 9, x = rem & (WW - 1);
        const float* a = A + (idx - rem);
        const float* b = Bp + (idx - rem);
        const float* f = F + (idx - rem);
        float gA = sobelAt(a, y, x), gB = sobelAt(b, y, x), gF = sobelAt(f, y, x);
        acc += fabsf(gF - fmaxf(gA, gB));
    }
    float t = blockReduce256(acc, sred);
    if (tid == 0) P[2 * 512 + blk] = t;
}

// ---------------- halo chain ----------------

__global__ void brightK(const float* __restrict__ A, const float* __restrict__ quant,
                        float* __restrict__ dst) {
    int idx = blockIdx.x * blockDim.x + threadIdx.x;
    if (idx >= NTOT) return;
    int img = idx >> 18;
    dst[idx] = (A[idx] > quant[img * 8 + 1]) ? 1.f : 0.f;
}

// vertical pass of exact disk-50 dilation: g(y,x) = min_{|dy|<=50, set} dy^2
__global__ void decayVK(const float* __restrict__ Ml, float* __restrict__ dst) {
    int idx = blockIdx.x * blockDim.x + threadIdx.x;
    if (idx >= NTOT) return;
    int rem = idx & (HWSZ - 1);
    int y = rem >> 9, x = rem & (WW - 1);
    const float* base = Ml + (idx - rem);
    float dmin = 1.0e30f;
    for (int dy = -50; dy <= 50; ++dy) {
        int yy = y + dy;
        int yyc = clampi(yy, 0, HH - 1);
        float v = base[yyc * WW + x];
        bool hit = ((unsigned)yy < (unsigned)HH) & (v > 0.f);
        float cand = hit ? (float)(dy * dy) : 1.0e30f;
        dmin = fminf(dmin, cand);
    }
    dst[idx] = dmin;
}

// horizontal pass + Mhalo + all halo/bloom partial sums
__global__ void spatialK(const float* __restrict__ bright, const float* __restrict__ dil,
                         const float* __restrict__ g, const float* __restrict__ Ml,
                         const float* __restrict__ M, const float* __restrict__ irb,
                         const float* __restrict__ visb, const float* __restrict__ fb,
                         float* __restrict__ P) {
    __shared__ float sred[256];
    int blk = blockIdx.x, tid = threadIdx.x;
    int base = blk * 2048;
    float aMh = 0.f, aHn = 0.f, aM = 0.f, aBn = 0.f;
    for (int i = tid; i < 2048; i += 256) {
        int idx = base + i;
        int rem = idx & (HWSZ - 1);
        int y = rem >> 9, x = rem & (WW - 1);
        const float* gimg = g + (idx - rem);
        float dmin = 1.0e30f;
        for (int dx = -50; dx <= 50; ++dx) {
            int xx = x + dx;
            int xxc = clampi(xx, 0, WW - 1);
            float gv = gimg[y * WW + xxc];
            bool inb = (unsigned)xx < (unsigned)WW;
            float cand = inb ? (gv + (float)(dx * dx)) : 1.0e30f;
            dmin = fminf(dmin, cand);
        }
        float decay = (dmin <= 2500.f) ? 1.f : 0.f;
        float halo = fminf(fmaxf(dil[idx] - bright[idx], 0.f), 1.f);
        float mh = fminf(fmaxf(halo - Ml[idx], 0.f), 1.f) * decay;
        aMh += mh;
        aHn += mh * fmaxf(fb[idx] - irb[idx], 0.f);
        float mv = M[idx];
        aM += mv;
        aBn += mv * fmaxf(fb[idx] - visb[idx], 0.f);
    }
    float t;
    t = blockReduce256(aMh, sred); if (tid == 0) P[3 * 512 + blk] = t;
    t = blockReduce256(aHn, sred); if (tid == 0) P[4 * 512 + blk] = t;
    t = blockReduce256(aM,  sred); if (tid == 0) P[5 * 512 + blk] = t;
    t = blockReduce256(aBn, sred); if (tid == 0) P[6 * 512 + blk] = t;
}

// ---------------- SSIM via WMMA (11-tap separable gaussian as banded matmul) ----------------
// One wave32 per 16x16 output tile. Inputs staged once into LDS (26x32 window of A,B,F,
// coalesced + branch-free borders). Per field: 2 horizontal WMMAs -> LDS f16 -> 1 vertical
// WMMA. 8 fields -> 24 v_wmma_f32_16x16x32_f16 per block.

#define TROWS 26
#define TCOLS 32

__global__ void __launch_bounds__(32) ssimK(const float* __restrict__ A,
                                            const float* __restrict__ Bp,
                                            const float* __restrict__ F,
                                            float* __restrict__ Pssim) {
    __shared__ float tile[3][TROWS * TCOLS];
    __shared__ _Float16 hbuf[32][16];
    __shared__ float red[32];
    int lane = threadIdx.x;
    int cx = blockIdx.x * 16;
    int ry = blockIdx.y * 16;
    int img = blockIdx.z;
    int n = lane & 15, halfId = lane >> 4;

    const float* srcA = A + (size_t)img * HWSZ;
    const float* srcB = Bp + (size_t)img * HWSZ;
    const float* srcF = F + (size_t)img * HWSZ;

    // ---- stage 26x32 window (rows ry-5..ry+20, cols cx-5..cx+26), zero-padded ----
    for (int i = lane; i < TROWS * TCOLS; i += 32) {
        int lr = i >> 5, lc = i & 31;
        int gy = ry - 5 + lr, gx = cx - 5 + lc;
        bool inb = ((unsigned)gy < (unsigned)HH) & ((unsigned)gx < (unsigned)WW);
        int off = clampi(gy, 0, HH - 1) * WW + clampi(gx, 0, WW - 1);
        float va = srcA[off], vb = srcB[off], vf = srcF[off];
        tile[0][i] = inb ? va : 0.f;
        tile[1][i] = inb ? vb : 0.f;
        tile[2][i] = inb ? vf : 0.f;
    }
    __syncthreads();

    // normalized gaussian (size 11, sigma 1.5)
    float gs = 0.f;
    #pragma unroll
    for (int t = 0; t < 11; ++t) { float d = (float)(t - 5); gs += expf(-d * d / 4.5f); }
    float invs = 1.f / gs;

    // banded-Toeplitz A fragment: A[m,k] = w[k-m] (0<=k-m<=10); identical for H and V passes
    U16h afr;
    {
        int m = n;
        #pragma unroll
        for (int e = 0; e < 16; ++e) {
            int j = e >> 1, p = e & 1;
            int k = halfId * 16 + 2 * j + p;
            int t = k - m;
            float w = 0.f;
            if (t >= 0 && t <= 10) { float d = (float)(t - 5); w = expf(-d * d / 4.5f) * invs; }
            afr.s[e] = (_Float16)w;
        }
    }

    U8f D[8];

    #pragma unroll
    for (int f = 0; f < 8; ++f) {
        // two horizontal conv tiles; D[m,nn] = h(row ry-5+16*t2+nn, col cx+m)
        #pragma unroll
        for (int t2 = 0; t2 < 2; ++t2) {
            int lr = 16 * t2 + n;               // local row in staged tile
            bool valid = lr < TROWS;            // rows >= 26 contribute zero weight anyway
            int lrc = valid ? lr : 0;
            const float* t0 = &tile[0][lrc * TCOLS];
            const float* t1 = &tile[1][lrc * TCOLS];
            const float* t2p = &tile[2][lrc * TCOLS];
            U16h bfr;
            #pragma unroll
            for (int e = 0; e < 16; ++e) {
                int lc = halfId * 16 + e;       // local col == k index of B
                float a = t0[lc], b = t1[lc], ff = t2p[lc];
                float v;
                switch (f) {
                    case 0: v = a;       break;
                    case 1: v = b;       break;
                    case 2: v = ff;      break;
                    case 3: v = a * a;   break;
                    case 4: v = b * b;   break;
                    case 5: v = ff * ff; break;
                    case 6: v = a * ff;  break;
                    default: v = b * ff; break;
                }
                v = valid ? v : 0.f;
                bfr.s[e] = (_Float16)v;
            }
            v8f cz = {};
            U8f du;
            du.v = __builtin_amdgcn_wmma_f32_16x16x32_f16(false, afr.v, false, bfr.v,
                                                          (short)0, cz, false, false);
            #pragma unroll
            for (int r = 0; r < 8; ++r) {
                int m = r + 8 * halfId;         // column within output tile
                hbuf[16 * t2 + n][m] = (_Float16)du.s[r];
            }
        }
        __syncthreads();
        // vertical conv tile: B[k,n] = hbuf[k][n]
        U16h bfr;
        #pragma unroll
        for (int e = 0; e < 16; ++e) {
            int k = halfId * 16 + e;
            bfr.s[e] = hbuf[k][n];
        }
        v8f cz = {};
        D[f].v = __builtin_amdgcn_wmma_f32_16x16x32_f16(false, afr.v, false, bfr.v,
                                                        (short)0, cz, false, false);
        __syncthreads();
    }

    // per-pixel SSIM maps for (A,F) and (B,F); lane owns 8 pixels (same m,n across fields)
    const float C1 = 1e-4f, C2 = 9e-4f;
    float sA = 0.f, sB = 0.f;
    #pragma unroll
    for (int r = 0; r < 8; ++r) {
        float muA = D[0].s[r], muB = D[1].s[r], muF = D[2].s[r];
        float eAA = D[3].s[r], eBB = D[4].s[r], eFF = D[5].s[r];
        float eAF = D[6].s[r], eBF = D[7].s[r];
        float vA = eAA - muA * muA, vB = eBB - muB * muB, vF = eFF - muF * muF;
        float cAF = eAF - muA * muF, cBF = eBF - muB * muF;
        sA += (2.f * muA * muF + C1) * (2.f * cAF + C2) /
              ((muA * muA + muF * muF + C1) * (vA + vF + C2));
        sB += (2.f * muB * muF + C1) * (2.f * cBF + C2) /
              ((muB * muB + muF * muF + C1) * (vB + vF + C2));
    }
    red[lane] = sA; __syncthreads();
    float tA = 0.f, tB = 0.f;
    if (lane == 0) { for (int i = 0; i < 32; ++i) tA += red[i]; }
    __syncthreads();
    red[lane] = sB; __syncthreads();
    if (lane == 0) {
        for (int i = 0; i < 32; ++i) tB += red[i];
        int blkId = blockIdx.x + blockIdx.y * 32 + blockIdx.z * 1024;
        Pssim[blkId] = tA;
        Pssim[4096 + blkId] = tB;
    }
}

// ---------------- final deterministic combine ----------------

__global__ void finalK(const float* __restrict__ P, const float* __restrict__ Pssim,
                       float* __restrict__ out) {
    if (threadIdx.x != 0 || blockIdx.x != 0) return;
    float l1 = 0.f, gr = 0.f, mh = 0.f, hn = 0.f, ms = 0.f, bn = 0.f;
    for (int i = 0; i < 512; ++i) {
        l1 += P[1 * 512 + i];
        gr += P[2 * 512 + i];
        mh += P[3 * 512 + i];
        hn += P[4 * 512 + i];
        ms += P[5 * 512 + i];
        bn += P[6 * 512 + i];
    }
    float sA = 0.f, sB = 0.f;
    for (int i = 0; i < 4096; ++i) { sA += Pssim[i]; sB += Pssim[4096 + i]; }
    float inv = 1.f / (float)NTOT;
    float loss_l1   = 15.f * l1 * inv;
    float loss_grad = 14.f * gr * inv;
    float loss_ssim = 20.f * (1.f - 0.5f * sA * inv - 0.5f * sB * inv);
    float loss_halo  = hn / fmaxf(mh + 1e-6f, 1.f);
    float loss_bloom = bn / fmaxf(ms + 1e-6f, 1.f);
    float whb = 0.3f * loss_halo + 0.6f * loss_bloom;
    out[0] = loss_l1 + loss_grad + loss_ssim + whb;
    out[1] = loss_grad;
    out[2] = loss_l1;
    out[3] = loss_ssim;
    out[4] = whb;
}

// ---------------- launch ----------------

extern "C" void kernel_launch(void* const* d_in, const int* in_sizes, int n_in,
                              void* d_out, int out_size, void* d_ws, size_t ws_size,
                              hipStream_t stream) {
    (void)in_sizes; (void)n_in; (void)out_size; (void)ws_size;
    const float* A  = (const float*)d_in[0];
    const float* Bp = (const float*)d_in[1];
    const float* F  = (const float*)d_in[2];
    float* out = (float*)d_out;
    float* ws  = (float*)d_ws;

    float* tmpA = ws;
    float* tmpB = ws + 1 * (size_t)NTOT;
    float* tmpC = ws + 2 * (size_t)NTOT;
    float* sal  = ws + 3 * (size_t)NTOT;
    float* M    = ws + 4 * (size_t)NTOT;
    float* Ml   = ws + 5 * (size_t)NTOT;
    float* irb  = ws + 6 * (size_t)NTOT;
    float* visb = ws + 7 * (size_t)NTOT;
    float* fb   = ws + 8 * (size_t)NTOT;
    float* scal = ws + 9 * (size_t)NTOT;
    float* quant   = scal;                 // 32 floats (img*8 + slot)
    float* salmean = scal + 32;            // 4 floats
    float* P       = scal + 64;            // 8 channels x 512 partials
    float* Pssim   = scal + 64 + 8 * 512;  // 2 x 4096 partials

    dim3 eb(256);
    dim3 eg(NTOT / 256);

    zeroK<<<52, 256, 0, stream>>>(scal, 64 + 8 * 512 + 8192);

    // quantiles of A and B
    quantK<<<dim3(5, 4), 256, 0, stream>>>(A, Bp, A, 0, quant);

    // saliency: sal = max_k (A - opening(A,k)), k in {9,15,31}
    int ks[3] = {9, 15, 31};
    int first = 1;
    for (int t = 0; t < 3; ++t) {
        int r = ks[t] / 2;
        poolK<0, 1><<<eg, eb, 0, stream>>>(A,    tmpA, r, 1.f, 0);   // erode H (min)
        poolK<0, 0><<<eg, eb, 0, stream>>>(tmpA, tmpB, r, 1.f, 0);   // erode V
        poolK<1, 1><<<eg, eb, 0, stream>>>(tmpB, tmpA, r, 1.f, 0);   // dilate H (max)
        poolK<1, 0><<<eg, eb, 0, stream>>>(tmpA, tmpB, r, 1.f, 0);   // dilate V
        salUpdateK<<<eg, eb, 0, stream>>>(A, tmpB, sal, first);
        first = 0;
    }
    chunkSumK<<<512, 256, 0, stream>>>(sal, P);                      // ch0 = raw sal sums
    salMeanK<<<1, 32, 0, stream>>>(P, salmean);
    salNormK<<<eg, eb, 0, stream>>>(sal, salmean);

    // quantiles of normalized sal
    quantK<<<dim3(2, 4), 256, 0, stream>>>(A, Bp, sal, 1, quant);

    computeMK<<<eg, eb, 0, stream>>>(sal, A, Bp, quant, M, Ml);
    l1K<<<512, 256, 0, stream>>>(A, Bp, F, M, quant, P);
    sobelK<<<512, 256, 0, stream>>>(A, Bp, F, P);

    // avgpool 9 (sum-pool, divide by 81; fused blur clipped to [0,1])
    poolK<2, 1><<<eg, eb, 0, stream>>>(A,    tmpA, 4, 1.f, 0);
    poolK<2, 0><<<eg, eb, 0, stream>>>(tmpA, irb,  4, 1.f / 81.f, 0);
    poolK<2, 1><<<eg, eb, 0, stream>>>(Bp,   tmpA, 4, 1.f, 0);
    poolK<2, 0><<<eg, eb, 0, stream>>>(tmpA, visb, 4, 1.f / 81.f, 0);
    poolK<2, 1><<<eg, eb, 0, stream>>>(F,    tmpA, 4, 1.f, 0);
    poolK<2, 0><<<eg, eb, 0, stream>>>(tmpA, fb,   4, 1.f / 81.f, 1);

    // halo mask + exact disk-50 decay + halo/bloom sums
    brightK<<<eg, eb, 0, stream>>>(A, quant, tmpA);                  // bright
    poolK<1, 1><<<eg, eb, 0, stream>>>(tmpA, tmpB, 7, 1.f, 0);       // dilate 15 H
    poolK<1, 0><<<eg, eb, 0, stream>>>(tmpB, tmpC, 7, 1.f, 0);       // dilate 15 V
    decayVK<<<eg, eb, 0, stream>>>(Ml, tmpB);                        // vertical dy^2 pass
    spatialK<<<512, 256, 0, stream>>>(tmpA, tmpC, tmpB, Ml, M, irb, visb, fb, P);

    // SSIM (WMMA), then finalize
    ssimK<<<dim3(32, 32, 4), 32, 0, stream>>>(A, Bp, F, Pssim);
    finalK<<<1, 32, 0, stream>>>(P, Pssim, out);
}
